// LowRankLayer_19954418057631
// MI455X (gfx1250) — matrix operand order repaired and checked by reading, not compiled, not run on hardware
//
#include <hip/hip_runtime.h>

// Problem sizes (compile-time constants from the reference)
#define BATCH   8
#define SEQ     2048
#define D_IN    4096
#define D_OUT   4096
#define RANK    256
#define M_TOT   (BATCH * SEQ)        // 16384 rows
#define KC      64                   // GEMM1 k-chunk (double-buffered in LDS)

typedef __bf16  v16bf __attribute__((ext_vector_type(16)));
typedef __bf16  v8bf  __attribute__((ext_vector_type(8)));
typedef __bf16  v4bf  __attribute__((ext_vector_type(4)));
typedef float   v8f   __attribute__((ext_vector_type(8)));
typedef float   v4f   __attribute__((ext_vector_type(4)));

// ---------------------------------------------------------------------------
// Kernel 0: fold s into V and transpose -> VsT[r][k] (bf16), convert U -> Ub
// ---------------------------------------------------------------------------
__global__ __launch_bounds__(256) void lowrank_prep(
    const float* __restrict__ U, const float* __restrict__ s,
    const float* __restrict__ V, __bf16* __restrict__ VsT,
    __bf16* __restrict__ Ub)
{
    int t = blockIdx.x * 256 + threadIdx.x;          // 0 .. 2*1M-1
    const int N1 = D_IN * RANK;                      // 1M
    if (t < N1) {
        int r = t >> 12;                             // t / 4096
        int k = t & (D_IN - 1);                      // t % 4096
        VsT[t] = (__bf16)(V[(size_t)k * RANK + r] * s[r]);
    } else {
        int u = t - N1;                              // o*RANK + r
        Ub[u] = (__bf16)U[u];
    }
}

// ---------------------------------------------------------------------------
// Kernel 1: T[m,r] = sum_k x[m,k] * Vs[k,r]   (M=16384, N=256, K=4096)
// One block = 16 rows x full N=256 (x read exactly once, NT).
// 8 waves, each wave -> 32 cols (2 tiles). Double-buffered LDS A tile,
// 64-wide k-chunks, one barrier per chunk, prefetch overlapped with WMMA.
// ---------------------------------------------------------------------------
__global__ __launch_bounds__(256) void lowrank_gemm1(
    const float* __restrict__ x, const __bf16* __restrict__ VsT,
    __bf16* __restrict__ T)
{
    const int m0    = blockIdx.x * 16;
    const int tid   = threadIdx.x;
    const int wave  = tid >> 5;
    const int lane  = tid & 31;
    const int l16   = lane & 15;
    const int lhalf = lane >> 4;          // 0 or 1
    const int n0    = wave * 32;          // wave's 2 output tiles: n0, n0+16

    __shared__ __bf16 As[2][16][72];      // 64 cols + 8 pad (conflict-free)

    v8f acc0 = {}; v8f acc1 = {};

    // staging coords: thread loads 4 consecutive f32 of one row (16B, 256B/row)
    const int sr = tid >> 4;              // row 0..15
    const int sc = (tid & 15) * 4;        // col 0..60
    const float* xp = x + (size_t)(m0 + sr) * D_IN + sc;

    // ---- prologue: stage chunk 0 into buffer 0
    {
        v4f f = __builtin_nontemporal_load((const v4f*)xp);
        v4bf pk;
        pk[0] = (__bf16)f[0]; pk[1] = (__bf16)f[1];
        pk[2] = (__bf16)f[2]; pk[3] = (__bf16)f[3];
        *(v4bf*)&As[0][sr][sc] = pk;
    }
    __syncthreads();

    const int NCHUNK = D_IN / KC;         // 64
    for (int i = 0; i < NCHUNK; ++i) {
        const int buf = i & 1;

        // ---- prefetch next chunk of x (issued before compute, NT)
        const int inext = (i + 1 < NCHUNK) ? (i + 1) : i;
        v4f fn = __builtin_nontemporal_load((const v4f*)(xp + (size_t)inext * KC));

        // ---- compute current chunk: 2 sub-steps of k=32, 4 WMMAs total
        #pragma unroll
        for (int sub = 0; sub < 2; ++sub) {
            const int ks = sub * 32;
            const __bf16* arow = &As[buf][l16][ks + lhalf * 16];
            v8bf alo = *(const v8bf*)arow;
            v8bf ahi = *(const v8bf*)(arow + 8);
            v16bf a = __builtin_shufflevector(alo, ahi,
                        0,1,2,3,4,5,6,7, 8,9,10,11,12,13,14,15);

            const int kb = i * KC + ks + lhalf * 16;
            v16bf b0 = *(const v16bf*)(VsT + (size_t)(n0 + l16) * D_IN + kb);
            v16bf b1 = *(const v16bf*)(VsT + (size_t)(n0 + 16 + l16) * D_IN + kb);

            acc0 = __builtin_amdgcn_wmma_f32_16x16x32_bf16(false, a, false, b0,
                                                           (short)0, acc0, false, false);
            acc1 = __builtin_amdgcn_wmma_f32_16x16x32_bf16(false, a, false, b1,
                                                           (short)0, acc1, false, false);
        }

        // ---- convert + store prefetched chunk into the other buffer
        {
            v4bf pk;
            pk[0] = (__bf16)fn[0]; pk[1] = (__bf16)fn[1];
            pk[2] = (__bf16)fn[2]; pk[3] = (__bf16)fn[3];
            *(v4bf*)&As[buf ^ 1][sr][sc] = pk;
        }
        __syncthreads();
    }

    // ---- store T as bf16: C VGPR v -> row v + lhalf*8, col l16
    #pragma unroll
    for (int v = 0; v < 8; ++v) {
        const int m = m0 + v + lhalf * 8;
        T[(size_t)m * RANK + n0 + l16]      = (__bf16)acc0[v];
        T[(size_t)m * RANK + n0 + 16 + l16] = (__bf16)acc1[v];
    }
}

// ---------------------------------------------------------------------------
// Kernel 2: out[m,o] = sum_r T[m,r] * Ub[o,r]  (M=16384, N=4096, K=256)
// Block tile 64x256 (8 waves as 4x2), wave tile 16x128 (8 accumulators).
// T and Ub are bf16, L2-resident, fragment-aligned -> no LDS, no barriers.
// Output is a 256MB write-once stream -> non-temporal stores.
// ---------------------------------------------------------------------------
__global__ __launch_bounds__(256) void lowrank_gemm2(
    const __bf16* __restrict__ T, const __bf16* __restrict__ Ub,
    float* __restrict__ out)
{
    const int tid   = threadIdx.x;
    const int wave  = tid >> 5;
    const int lane  = tid & 31;
    const int l16   = lane & 15;
    const int lhalf = lane >> 4;
    const int wm = wave >> 1;                       // 0..3
    const int wn = wave & 1;                        // 0..1
    const int m0    = blockIdx.x * 64 + wm * 16;
    const int nbase = blockIdx.y * 256 + wn * 128;

    v8f acc[8];
    #pragma unroll
    for (int j = 0; j < 8; ++j) acc[j] = (v8f){};

    const __bf16* arowp = T + (size_t)(m0 + l16) * RANK;

    #pragma unroll
    for (int k0 = 0; k0 < RANK; k0 += 32) {
        // A fragment: row m0+l16, k runs at k0+lhalf*16 .. +15
        const int abase = k0 + lhalf * 16;
        v8bf alo = *(const v8bf*)(arowp + abase);
        v8bf ahi = *(const v8bf*)(arowp + abase + 8);
        v16bf a = __builtin_shufflevector(alo, ahi,
                    0,1,2,3,4,5,6,7, 8,9,10,11,12,13,14,15);

        const int kb = k0 + lhalf * 16;
        #pragma unroll
        for (int j = 0; j < 8; ++j) {
            v16bf b = *(const v16bf*)(Ub + (size_t)(nbase + j * 16 + l16) * RANK + kb);
            acc[j] = __builtin_amdgcn_wmma_f32_16x16x32_bf16(false, a, false, b,
                                                             (short)0, acc[j], false, false);
        }
    }

    // epilogue: f32 NT stores, 64B contiguous per half-wave per VGPR row
    #pragma unroll
    for (int v = 0; v < 8; ++v) {
        const int m = m0 + v + lhalf * 8;
        float* orow = out + (size_t)m * D_OUT + nbase + l16;
        #pragma unroll
        for (int j = 0; j < 8; ++j)
            __builtin_nontemporal_store(acc[j][v], &orow[j * 16]);
    }
}

// ---------------------------------------------------------------------------
// Launcher. Inputs: x[8,2048,4096] f32, u[4096,256] f32, s[256] f32,
// v[4096,256] f32. Output: [8,2048,4096] f32. Workspace: 12 MB used.
// ---------------------------------------------------------------------------
extern "C" void kernel_launch(void* const* d_in, const int* in_sizes, int n_in,
                              void* d_out, int out_size, void* d_ws, size_t ws_size,
                              hipStream_t stream)
{
    const float* x = (const float*)d_in[0];
    const float* U = (const float*)d_in[1];
    const float* s = (const float*)d_in[2];
    const float* V = (const float*)d_in[3];
    float* out = (float*)d_out;

    char* ws = (char*)d_ws;
    __bf16* VsT = (__bf16*)(ws);                                   // 2 MB [256][4096]
    __bf16* Ub  = (__bf16*)(ws + (size_t)RANK * D_IN * 2);         // 2 MB [4096][256]
    __bf16* T   = (__bf16*)(ws + (size_t)RANK * D_IN * 2
                               + (size_t)D_OUT * RANK * 2);        // 8 MB [16384][256]

    // 1) fold scale + transpose V, convert U   (2M elements)
    lowrank_prep<<<(2 * D_IN * RANK) / 256, 256, 0, stream>>>(U, s, V, VsT, Ub);

    // 2) T = x @ Vs   (blocks: one per 16-row tile, full N per block)
    lowrank_gemm1<<<M_TOT / 16, 256, 0, stream>>>(x, VsT, T);

    // 3) out = T @ Ub^T   (64x256 block tiles)
    lowrank_gemm2<<<dim3(M_TOT / 64, D_OUT / 256), 256, 0, stream>>>(T, Ub, out);
}
